// Model_68324339744905
// MI455X (gfx1250) — matrix operand order repaired and checked by reading, not compiled, not run on hardware
//
#include <hip/hip_runtime.h>

#define NN 100000
#define EE 1200000
#define DH 64
#define DOUT 40

typedef __attribute__((ext_vector_type(16))) __bf16 v16bf;
typedef __attribute__((ext_vector_type(8)))  float  v8f;

// ---------------------------------------------------------------------------
// Wave (32-lane) sum reduction
// ---------------------------------------------------------------------------
__device__ __forceinline__ float wave_sum(float v) {
#pragma unroll
    for (int o = 16; o > 0; o >>= 1) v += __shfl_xor(v, o, 32);
    return v;
}

// ---------------------------------------------------------------------------
// Degree / aggregation kernels  (D^-1/2 (A+I) D^-1/2; aggregation moved
// BEFORE the weight multiply -- linearity of segment_sum)
// ---------------------------------------------------------------------------
__global__ void fill_ones(float* deg, int n) {
    int i = blockIdx.x * 256 + threadIdx.x;
    if (i < n) deg[i] = 1.0f;              // self-loop contributes 1 to degree
}

__global__ void count_deg(const int* __restrict__ dst, float* __restrict__ deg, int ne) {
    int e = blockIdx.x * 256 + threadIdx.x;
    if (e < ne) unsafeAtomicAdd(&deg[dst[e]], 1.0f);
}

__global__ void finish_dinv(float* deg, int n) {
    int i = blockIdx.x * 256 + threadIdx.x;
    if (i < n) deg[i] = rsqrtf(deg[i]);    // deg >= 1 always (self-loop)
}

// out[i,:] = feat[i,:] * dinv[i]^2   (the self-loop term), full overwrite
__global__ void selfloop_init(const float* __restrict__ feat, const float* __restrict__ dinv,
                              float* __restrict__ out, int n) {
    int i = blockIdx.x * 256 + threadIdx.x;
    if (i < n * DH) {
        float d = dinv[i >> 6];
        out[i] = feat[i] * d * d;
    }
}

// one wave per edge, float2 per lane (64 floats)
__global__ void agg_edges(const int* __restrict__ src, const int* __restrict__ dst,
                          const float* __restrict__ dinv, const float* __restrict__ feat,
                          float* __restrict__ out, int ne) {
    int lane = threadIdx.x & 31;
    int e = blockIdx.x * 8 + (threadIdx.x >> 5);
    if (e >= ne) return;
    int s = src[e], d = dst[e];
    float c = dinv[s] * dinv[d];
    const float2 v = ((const float2*)(feat + (size_t)s * DH))[lane];
    float* o = out + (size_t)d * DH + lane * 2;
    unsafeAtomicAdd(o,     v.x * c);
    unsafeAtomicAdd(o + 1, v.y * c);
}

// ---------------------------------------------------------------------------
// WMMA GEMM: C[M x ncols] = A[M x 64] @ W + bias, per-head weight blocks.
//   W is groups of [64 x head_cols] row-major; bias is flat over columns.
//   mode: 0 = none, 1 = relu.
//   Each wave computes a 16x64 strip: A fragments loaded ONCE (8x b128 +
//   cvt_pk_bf16), then 4 column tiles x 2 wmma(K=32) with B fragments
//   pre-staged in LDS in exact WMMA per-lane order (2x ds_load_b128 each).
//   grid.y covers 64-column groups; out-of-range columns are zero-padded at
//   staging (clamped addresses -> never OOB) and masked at store.
// ---------------------------------------------------------------------------
__global__ __launch_bounds__(256)
void gemm_wmma(const float* __restrict__ A, int lda,
               const float* __restrict__ W, const float* __restrict__ bias,
               float* __restrict__ C, int ldc,
               int M, int ncols, int head_cols, int mode) {
    const int lane = threadIdx.x & 31;
    const int wave = threadIdx.x >> 5;
    const int g0   = blockIdx.y * 64;                  // first column of group
    const int half = lane >> 4;
    const int l15  = lane & 15;

    // ---- stage B strip into LDS in WMMA fragment order: [tile][kstep][lane][elem]
    // elem e of lane L (kstep s): k = s*32 + (e>=8)*16 + (L>>4)*8 + (e&7)
    __shared__ __align__(16) __bf16 Bs[4096];          // 4 tiles * 2 ksteps * 32 * 16
    for (int f = threadIdx.x; f < 4096; f += 256) {
        int t  = f >> 10;
        int s  = (f >> 9) & 1;
        int e  = f & 15;
        int L  = (f >> 4) & 31;
        int k  = s * 32 + ((e >> 3) << 4) + ((L >> 4) << 3) + (e & 7);
        int gn = g0 + t * 16 + (L & 15);
        int gc = (gn < ncols) ? gn : (ncols - 1);      // clamp: address always valid
        int hd = gc / head_cols;
        int nh = gc - hd * head_cols;
        float v = (gn < ncols) ? W[hd * 64 * head_cols + k * head_cols + nh] : 0.0f;
        Bs[f] = (__bf16)v;
    }
    __syncthreads();

    const int m0 = (blockIdx.x * 8 + wave) * 16;
    if (m0 >= M) return;                               // wave-uniform exit

    const int mrow = m0 + l15;                         // lane's A row (M % 16 == 0)
    const float* Ar = A + mrow * lda;

    // ---- A fragments for both K-steps, loaded once for the whole strip
    v16bf a01[2];
#pragma unroll
    for (int s = 0; s < 2; ++s) {
        const int kb = s * 32 + half * 8;              // two contiguous 8-float runs
        float4 p0 = *(const float4*)(Ar + kb);
        float4 p1 = *(const float4*)(Ar + kb + 4);
        float4 p2 = *(const float4*)(Ar + kb + 16);
        float4 p3 = *(const float4*)(Ar + kb + 20);
        v16bf a;
        a[0]  = (__bf16)p0.x; a[1]  = (__bf16)p0.y; a[2]  = (__bf16)p0.z; a[3]  = (__bf16)p0.w;
        a[4]  = (__bf16)p1.x; a[5]  = (__bf16)p1.y; a[6]  = (__bf16)p1.z; a[7]  = (__bf16)p1.w;
        a[8]  = (__bf16)p2.x; a[9]  = (__bf16)p2.y; a[10] = (__bf16)p2.z; a[11] = (__bf16)p2.w;
        a[12] = (__bf16)p3.x; a[13] = (__bf16)p3.y; a[14] = (__bf16)p3.z; a[15] = (__bf16)p3.w;
        a01[s] = a;
    }

    // ---- 4 column tiles: 4 ds_load_b128 + 2 wmma + fused epilogue each
#pragma unroll
    for (int t = 0; t < 4; ++t) {
        const v16bf b0 = *reinterpret_cast<const v16bf*>(&Bs[t * 1024 + (lane << 4)]);
        const v16bf b1 = *reinterpret_cast<const v16bf*>(&Bs[t * 1024 + 512 + (lane << 4)]);
        v8f acc = {};
        acc = __builtin_amdgcn_wmma_f32_16x16x32_bf16(false, a01[0], false, b0, (short)0, acc, false, false);
        acc = __builtin_amdgcn_wmma_f32_16x16x32_bf16(false, a01[1], false, b1, (short)0, acc, false, false);

        const int gn = g0 + t * 16 + l15;              // this lane's output column
        const bool nok = (gn < ncols);
        const float bv = nok ? bias[gn] : 0.0f;        // heads contiguous -> flat bias
#pragma unroll
        for (int r = 0; r < 8; ++r) {
            int m = m0 + r + half * 8;                 // C/D layout: VGPR r -> row
            if (nok) {
                float v = acc[r] + bv;
                if (mode == 1) v = fmaxf(v, 0.0f);
                C[m * ldc + gn] = v;
            }
        }
    }
}

// ---------------------------------------------------------------------------
// Gated multi-head combine + residual + layernorm.  One wave per node,
// 2 channels per lane.  Hcat==nullptr -> hk is broadcast bias (layer 0, h=0).
// ---------------------------------------------------------------------------
__global__ void gate_ln(const float* __restrict__ Hcat, const float* __restrict__ hbias,
                        const float* __restrict__ Xcat,
                        const float* __restrict__ gfw, const float* __restrict__ gfb,
                        const float* __restrict__ guw, const float* __restrict__ gub,
                        const float* __restrict__ hprev,
                        float* __restrict__ out, int n) {
    int lane = threadIdx.x & 31;
    int node = blockIdx.x * 8 + (threadIdx.x >> 5);
    if (node >= n) return;
    const int c0 = lane * 2, c1 = c0 + 1;
    const float fb = gfb[0], ub = gub[0];
    float o0 = 0.0f, o1 = 0.0f;
#pragma unroll
    for (int k = 0; k < 4; ++k) {
        const float* hr = Hcat ? (Hcat + (size_t)node * 256 + k * 64) : (hbias + k * 64);
        const float* xr = Xcat + (size_t)node * 256 + k * 64;
        float h0 = hr[c0], h1 = hr[c1];
        float x0 = xr[c0], x1 = xr[c1];
        float fd = h0 * gfw[c0] + h1 * gfw[c1] + x0 * gfw[64 + c0] + x1 * gfw[64 + c1];
        float ud = h0 * guw[c0] + h1 * guw[c1] + x0 * guw[64 + c0] + x1 * guw[64 + c1];
        fd = wave_sum(fd);
        ud = wave_sum(ud);
        float f = 0.2f * tanhf(fd + fb);
        float u = 0.2f * tanhf(ud + ub);
        o0 += (1.0f + f) * h0 + (1.0f + u) * x0;
        o1 += (1.0f + f) * h1 + (1.0f + u) * x1;
    }
    o0 *= 0.25f; o1 *= 0.25f;
    if (hprev) {
        o0 += hprev[(size_t)node * DH + c0];
        o1 += hprev[(size_t)node * DH + c1];
    }
    float mu = wave_sum(o0 + o1) * (1.0f / 64.0f);
    float d0 = o0 - mu, d1 = o1 - mu;
    float var = wave_sum(d0 * d0 + d1 * d1) * (1.0f / 64.0f);
    float rs = rsqrtf(var + 1e-5f);
    out[(size_t)node * DH + c0] = d0 * rs;
    out[(size_t)node * DH + c1] = d1 * rs;
}

// out = LayerNorm(a + b), one wave per node
__global__ void add_ln(const float* __restrict__ a, const float* __restrict__ b,
                       float* __restrict__ out, int n) {
    int lane = threadIdx.x & 31;
    int node = blockIdx.x * 8 + (threadIdx.x >> 5);
    if (node >= n) return;
    size_t base = (size_t)node * DH;
    int c0 = lane * 2, c1 = c0 + 1;
    float v0 = a[base + c0] + b[base + c0];
    float v1 = a[base + c1] + b[base + c1];
    float mu = wave_sum(v0 + v1) * (1.0f / 64.0f);
    float d0 = v0 - mu, d1 = v1 - mu;
    float var = wave_sum(d0 * d0 + d1 * d1) * (1.0f / 64.0f);
    float rs = rsqrtf(var + 1e-5f);
    out[base + c0] = d0 * rs;
    out[base + c1] = d1 * rs;
}

// ---------------------------------------------------------------------------
extern "C" void kernel_launch(void* const* d_in, const int* in_sizes, int n_in,
                              void* d_out, int out_size, void* d_ws, size_t ws_size,
                              hipStream_t stream) {
    const float* x    = (const float*)d_in[0];
    const int*   ei   = (const int*)  d_in[1];
    const float* Wh   = (const float*)d_in[2];
    const float* bh   = (const float*)d_in[3];
    const float* Wx   = (const float*)d_in[4];
    const float* bx   = (const float*)d_in[5];
    const float* gfw  = (const float*)d_in[6];
    const float* gfb  = (const float*)d_in[7];
    const float* guw  = (const float*)d_in[8];
    const float* gub  = (const float*)d_in[9];
    const float* ff1w = (const float*)d_in[10];
    const float* ff1b = (const float*)d_in[11];
    const float* ff2w = (const float*)d_in[12];
    const float* ff2b = (const float*)d_in[13];
    const float* c1w  = (const float*)d_in[14];
    const float* c1b  = (const float*)d_in[15];
    const float* c2w  = (const float*)d_in[16];
    const float* c2b  = (const float*)d_in[17];
    float* out = (float*)d_out;

    const int* esrc = ei;
    const int* edst = ei + EE;

    // workspace carve-up (fp32), ~308 MB total
    float* p    = (float*)d_ws;
    float* dinv = p;  p += NN;                 // degrees -> dinv
    float* Ax   = p;  p += (size_t)NN * DH;    // aggregated x (reused both layers)
    float* Ah   = p;  p += (size_t)NN * DH;    // aggregated h; also FF2 output
    float* Xcat = p;  p += (size_t)NN * 256;   // per-head x-conv outputs; also FF/clf temp
    float* Hcat = p;  p += (size_t)NN * 256;   // per-head h-conv outputs
    float* h    = p;  p += (size_t)NN * DH;    // block output
    float* hmid = p;  p += (size_t)NN * DH;    // post-gate pre-FF
    float* t0   = Xcat;                        // alias: free while FF/classifier runs

    const dim3 b256(256);
    const int gN    = (NN + 255) / 256;
    const int gNDH  = (NN * DH + 255) / 256;
    const int gE    = (EE + 255) / 256;
    const int gEw   = (EE + 7) / 8;            // wave-per-edge
    const int gNw   = (NN + 7) / 8;            // wave-per-node
    const int gM    = (NN / 16 + 7) / 8;       // 6250 row tiles / 8 waves

    // ---- degree / normalization --------------------------------------------
    fill_ones  <<<gN,   b256, 0, stream>>>(dinv, NN);
    count_deg  <<<gE,   b256, 0, stream>>>(edst, dinv, EE);
    finish_dinv<<<gN,   b256, 0, stream>>>(dinv, NN);

    // ---- Ax = D^-1/2 (A+I) D^-1/2 x  (aggregate once, reuse everywhere) ----
    selfloop_init<<<gNDH, b256, 0, stream>>>(x, dinv, Ax, NN);
    agg_edges    <<<gEw,  b256, 0, stream>>>(esrc, edst, dinv, x, Ax, EE);

    // ======================= layer 0 (h = 0) ================================
    gemm_wmma<<<dim3(gM, 4), b256, 0, stream>>>(Ax, DH, Wx,        bx,        Xcat, 256, NN, 256, 64, 0);
    gate_ln  <<<gNw, b256, 0, stream>>>(nullptr, bh, Xcat, gfw, gfb, guw, gub, nullptr, hmid, NN);
    gemm_wmma<<<dim3(gM, 1), b256, 0, stream>>>(hmid, DH, ff1w,      ff1b,      t0, DH, NN, 64, 64, 1);
    gemm_wmma<<<dim3(gM, 1), b256, 0, stream>>>(t0,   DH, ff2w,      ff2b,      Ah, DH, NN, 64, 64, 0);
    add_ln   <<<gNw, b256, 0, stream>>>(Ah, hmid, h, NN);

    // ---- Ah = aggregate(h) --------------------------------------------------
    selfloop_init<<<gNDH, b256, 0, stream>>>(h, dinv, Ah, NN);
    agg_edges    <<<gEw,  b256, 0, stream>>>(esrc, edst, dinv, h, Ah, EE);

    // ======================= layer 1 ========================================
    gemm_wmma<<<dim3(gM, 4), b256, 0, stream>>>(Ah, DH, Wh + 16384, bh + 256,  Hcat, 256, NN, 256, 64, 0);
    gemm_wmma<<<dim3(gM, 4), b256, 0, stream>>>(Ax, DH, Wx + 16384, bx + 256,  Xcat, 256, NN, 256, 64, 0);
    gate_ln  <<<gNw, b256, 0, stream>>>(Hcat, bh, Xcat, gfw, gfb, guw, gub, h, hmid, NN);
    gemm_wmma<<<dim3(gM, 1), b256, 0, stream>>>(hmid, DH, ff1w + 4096, ff1b + 64, t0, DH, NN, 64, 64, 1);
    gemm_wmma<<<dim3(gM, 1), b256, 0, stream>>>(t0,   DH, ff2w + 4096, ff2b + 64, Ah, DH, NN, 64, 64, 0);
    add_ln   <<<gNw, b256, 0, stream>>>(Ah, hmid, h, NN);

    // ======================= classifier =====================================
    gemm_wmma<<<dim3(gM, 1), b256, 0, stream>>>(h,  DH, c1w, c1b, t0,  DH,   NN, 64, 64, 1);
    gemm_wmma<<<dim3(gM, 1), b256, 0, stream>>>(t0, DH, c2w, c2b, out, DOUT, NN, DOUT, DOUT, 0);
}
